// ConditionalRBM_56994216018034
// MI455X (gfx1250) — matrix operand order repaired and checked by reading, compiled, not run on hardware
//
#include <hip/hip_runtime.h>

typedef __attribute__((ext_vector_type(16))) _Float16 v16h;
typedef __attribute__((ext_vector_type(8)))  _Float16 v8h;
typedef __attribute__((ext_vector_type(8)))  float    v8f;

#define BATCH     1024
#define NN        256          // num_u == num_v == num_h
#define TILE_ROWS 16
#define THREADS   512          // 16 wave32 -> 4 waves per SIMD32, one nt tile each
#define THERM     100
#define REC       32
#define STEPS     (THERM + REC)
#define NV        (BATCH * NN)
#define NH        (BATCH * NN)
#define WB_ELEMS  (NN * NN)    // 65536 halfs per swizzled B buffer

__device__ __forceinline__ float sigmoidf_(float x) {
    return 1.0f / (1.0f + __expf(-x));
}

// One workgroup (512 thr = 16 wave32) owns TILE_ROWS=16 batch rows for all
// 132 Gibbs steps; each wave owns one 16-column output tile for both GEMMs.
// Weights live in LDS (fp16, pre-swizzled to the WMMA B-operand lane layout),
// v/h tiles live in LDS; u2h/u2v are hoisted into registers and fed as the
// WMMA C operand. Only the uniform stream touches HBM inside the loop.
extern "C" __global__ __launch_bounds__(THREADS, 1)
void crbm_gibbs_kernel(const float* __restrict__ u_state,    // [B, 256]
                       const float* __restrict__ w_vu,       // [256, 256]
                       const float* __restrict__ w_hu,       // [256, 256]
                       const float* __restrict__ w_hv,       // [256, 256]
                       const float* __restrict__ bias_v,     // [256]
                       const float* __restrict__ bias_h,     // [256]
                       const float* __restrict__ uniform,    // [69,468,160]
                       float* __restrict__ out)              // [32, 1024, 256]
{
    extern __shared__ char smem[];
    _Float16* Bt = (_Float16*)smem;        // W_hv^T swizzled: ph GEMM  (128 KB)
    _Float16* Bn = Bt + WB_ELEMS;          // W_hv   swizzled: pv GEMM  (128 KB)
    _Float16* vbuf = Bn + WB_ELEMS;        // v tile [16][256] fp16     (  8 KB)
    _Float16* hbuf = vbuf + TILE_ROWS * NN;// h tile [16][256] fp16     (  8 KB)
    float* u2h = (float*)(hbuf + TILE_ROWS * NN);   // [16][256] f32    ( 16 KB)
    float* u2v = u2h + TILE_ROWS * NN;              // [16][256] f32    ( 16 KB)

    const int tid  = threadIdx.x;
    const int row0 = blockIdx.x * TILE_ROWS;   // first global batch row

    // ---- Stage W_hv into LDS twice, fp16, pre-swizzled to WMMA-B layout ----
    // B-operand layout (16-bit, 32x16): lane L holds column N = L&15,
    // K = (L>>4)*8 + (i&7) + (i>=8?16:0) for i = 0..15 packed (one 32B read/step).
    // Buffer index = (((kc*16 + nt)*32 + lane)*16 + i).
    for (int idx = tid; idx < WB_ELEMS; idx += THREADS) {
        int i    = idx & 15;
        int lane = (idx >> 4) & 31;
        int nt   = (idx >> 9) & 15;
        int kc   = idx >> 13;
        int K  = kc * 32 + (lane >> 4) * 8 + (i & 7) + ((i >> 3) << 4);
        int Nc = nt * 16 + (lane & 15);
        Bt[idx] = (_Float16)w_hv[Nc * NN + K];   // (W_hv^T)[K][Nc] = W_hv[Nc][K]
        Bn[idx] = (_Float16)w_hv[K * NN + Nc];   //  W_hv  [K][Nc]
    }

    // ---- One-time per-row constants (exact fp32) + initial visible state ----
    for (int e = tid; e < TILE_ROWS * NN; e += THREADS) {
        int m = e >> 8, n = e & 255;
        const float* ur  = u_state + (size_t)(row0 + m) * NN;
        const float* whr = w_hu + n * NN;
        const float* wvr = w_vu + n * NN;
        float sh = bias_h[n], sv = bias_v[n];
        for (int k = 0; k < NN; ++k) {
            float uk = ur[k];
            sh += uk * whr[k];
            sv += uk * wvr[k];
        }
        u2h[e] = sh;
        u2v[e] = sv;
        float p0 = sigmoidf_(sv);
        vbuf[e] = (uniform[(size_t)(row0 + m) * NN + n] < p0)
                      ? (_Float16)1.0f : (_Float16)0.0f;
    }
    __syncthreads();

    const int wv   = tid >> 5;     // wave id 0..15 == column tile nt
    const int lane = tid & 31;
    const int hh   = lane >> 4;    // lane half (0/1)
    const int cn   = lane & 15;    // A row index / C column-within-tile
    const int nt   = wv;
    const int ncol = nt * 16 + cn; // this lane's output column

    // Step-invariant conditioning terms -> registers (C operand seeds).
    float c2h[8], c2v[8];
    #pragma unroll
    for (int r = 0; r < 8; ++r) {
        int m = hh * 8 + r;
        c2h[r] = u2h[m * NN + ncol];
        c2v[r] = u2v[m * NN + ncol];
    }

    const _Float16* Bt_base = Bt + ((nt * 32 + lane) << 4);  // + kc*8192
    const _Float16* Bn_base = Bn + ((nt * 32 + lane) << 4);

    for (int s = 0; s < STEPS; ++s) {
        const float* rh = uniform + NV + (size_t)s * (NH + NV); // step s, h draws
        const float* rv = rh + NH;                              // step s, v draws

        // Prefetch next step's uniforms: each wave covers exactly its own
        // (row, column-tile) cachelines.
        if (s + 1 < STEPS) {
            const float* rnext = rh + (NH + NV);
            #pragma unroll
            for (int r = 0; r < 8; ++r) {
                size_t off = (size_t)(row0 + hh * 8 + r) * NN + nt * 16;
                __builtin_prefetch(rnext + off, 0, 1);
                __builtin_prefetch(rnext + NH + off, 0, 1);
            }
        }

        // ---------------- GEMM1: ph = v @ W_hv^T + u2h ----------------
        v8f acc;
        #pragma unroll
        for (int r = 0; r < 8; ++r) acc[r] = c2h[r];
        #pragma unroll
        for (int kc = 0; kc < 8; ++kc) {
            // A (16x32 f16): lane holds row cn; K = kc*32 + hh*8 + (i&7) + (i>=8?16:0)
            const _Float16* ap = vbuf + cn * NN + kc * 32 + hh * 8;
            v8h lo = *(const v8h*)ap;
            v8h hi = *(const v8h*)(ap + 16);
            v16h a;
            #pragma unroll
            for (int i = 0; i < 8; ++i) { a[i] = lo[i]; a[i + 8] = hi[i]; }
            v16h b = *(const v16h*)(Bt_base + kc * 8192);
            acc = __builtin_amdgcn_wmma_f32_16x16x32_f16(false, a, false, b,
                                                         (short)0, acc, false, false);
        }
        // C layout: acc[r] -> row m = hh*8 + r, col = ncol
        #pragma unroll
        for (int r = 0; r < 8; ++r) {
            int m  = hh * 8 + r;
            int gb = row0 + m;
            float p = sigmoidf_(acc[r]);
            hbuf[m * NN + ncol] = (rh[(size_t)gb * NN + ncol] < p)
                                      ? (_Float16)1.0f : (_Float16)0.0f;
        }
        __syncthreads();   // all h columns visible to all waves

        // ---------------- GEMM2: pv = h @ W_hv + u2v ----------------
        v8f d;
        #pragma unroll
        for (int r = 0; r < 8; ++r) d[r] = c2v[r];
        #pragma unroll
        for (int kc = 0; kc < 8; ++kc) {
            const _Float16* ap = hbuf + cn * NN + kc * 32 + hh * 8;
            v8h lo = *(const v8h*)ap;
            v8h hi = *(const v8h*)(ap + 16);
            v16h a;
            #pragma unroll
            for (int i = 0; i < 8; ++i) { a[i] = lo[i]; a[i + 8] = hi[i]; }
            v16h b = *(const v16h*)(Bn_base + kc * 8192);
            d = __builtin_amdgcn_wmma_f32_16x16x32_f16(false, a, false, b,
                                                       (short)0, d, false, false);
        }
        #pragma unroll
        for (int r = 0; r < 8; ++r) {
            int m  = hh * 8 + r;
            int gb = row0 + m;
            float p = sigmoidf_(d[r]);
            bool bit = rv[(size_t)gb * NN + ncol] < p;
            vbuf[m * NN + ncol] = bit ? (_Float16)1.0f : (_Float16)0.0f;
            if (s >= THERM) {
                out[((size_t)(s - THERM) * BATCH + gb) * NN + ncol] = bit ? 1.0f : 0.0f;
            }
        }
        __syncthreads();   // new v visible before next step's GEMM1
    }
}

extern "C" void kernel_launch(void* const* d_in, const int* in_sizes, int n_in,
                              void* d_out, int out_size, void* d_ws, size_t ws_size,
                              hipStream_t stream) {
    (void)in_sizes; (void)n_in; (void)d_ws; (void)ws_size; (void)out_size;
    const float* u   = (const float*)d_in[0];
    const float* wvu = (const float*)d_in[1];
    const float* whu = (const float*)d_in[2];
    const float* whv = (const float*)d_in[3];
    const float* bv  = (const float*)d_in[4];
    const float* bh  = (const float*)d_in[5];
    const float* un  = (const float*)d_in[6];
    float* out = (float*)d_out;

    // LDS: 2*128KB swizzled W + 2*8KB v/h tiles + 2*16KB u2h/u2v = 304 KB
    size_t shmem = (size_t)(2 * WB_ELEMS) * sizeof(_Float16)
                 + (size_t)(2 * TILE_ROWS * NN) * sizeof(_Float16)
                 + (size_t)(2 * TILE_ROWS * NN) * sizeof(float);

    crbm_gibbs_kernel<<<dim3(BATCH / TILE_ROWS), dim3(THREADS), shmem, stream>>>(
        u, wvu, whu, whv, bv, bh, un, out);
}
